// TestRNN_65850438582490
// MI455X (gfx1250) — compile-verified, matrix operand order checked by reading, not run on hardware
//
#include <hip/hip_runtime.h>

// ---------------------------------------------------------------------------
// 4-step LSTM cell (B=2M, D=H=32) on gfx1250 using bf16 WMMA (f32 accumulate).
// One wave32 owns a 16-row batch tile; block = 8 waves = 128 rows.
// Memory-bound target: ~1 GiB streaming traffic -> ~46us at 23.3 TB/s.
// Activations use v_tanh_f32 for BOTH tanh and sigmoid (s(x)=0.5*tanh(x/2)+0.5)
// => 5 trans ops per gate element per step instead of 8.
// ---------------------------------------------------------------------------

typedef __bf16 bf16_t;
typedef __attribute__((ext_vector_type(16))) __bf16 v16bf;
typedef __attribute__((ext_vector_type(8)))  __bf16 v8bf;
typedef __attribute__((ext_vector_type(8)))  float  v8f;
typedef __attribute__((ext_vector_type(4)))  float  v4f;

#define LOG2E     1.44269504088896340736f
#define NEG_SLOPE 0.01f
#define HPAD      40   // padded bf16 row stride for the h-transpose scratch

__device__ __forceinline__ float fast_tanh(float x) {
#if __has_builtin(__builtin_amdgcn_tanhf)
  return __builtin_amdgcn_tanhf(x);            // v_tanh_f32 (confirmed on gfx1250)
#else
  float e = exp2f(-2.0f * LOG2E * x);
  return (1.0f - e) / (1.0f + e);
#endif
}
__device__ __forceinline__ float sigmoidf_fast(float x) {
#if __has_builtin(__builtin_amdgcn_tanhf)
  // sigma(x) = 0.5*tanh(x/2) + 0.5  -> single v_tanh_f32 + mul + fma
  return __builtin_fmaf(0.5f, __builtin_amdgcn_tanhf(0.5f * x), 0.5f);
#else
#if __has_builtin(__builtin_amdgcn_exp2f) && __has_builtin(__builtin_amdgcn_rcpf)
  return __builtin_amdgcn_rcpf(1.0f + __builtin_amdgcn_exp2f(-LOG2E * x));
#else
  return 1.0f / (1.0f + exp2f(-LOG2E * x));
#endif
#endif
}

__device__ __forceinline__ v8f wmma_bf16(v16bf a, v16bf b, v8f c) {
  // D = A(16x32 bf16) * B(32x16 bf16) + C(16x16 f32)
  return __builtin_amdgcn_wmma_f32_16x16x32_bf16(
      /*neg_a=*/false, a, /*neg_b=*/false, b,
      /*c_mod=*/(short)0, c, /*reuse_a=*/false, /*reuse_b=*/false);
}

__global__ __launch_bounds__(256)
void lstm4_wmma_kernel(const float* __restrict__ X,    // [B,32] token_embedding
                       const float* __restrict__ H0,   // [B,32]
                       const float* __restrict__ C0,   // [B,32]
                       const float* __restrict__ Wih,  // [128,32]
                       const float* __restrict__ Whh,  // [128,32]
                       const float* __restrict__ bih,  // [128]
                       const float* __restrict__ bhh,  // [128]
                       float* __restrict__ out,        // [B,32]
                       long B) {
  // t = 0..7  : W_ih^T 32x16 column tiles (bf16 B fragments)
  // t = 8..15 : W_hh^T 32x16 column tiles
  __shared__ v16bf  wfrag[16][32];            // 16 KB
  __shared__ float  bias[128];                // b_ih + b_hh
  __shared__ bf16_t hscr[8][16][HPAD];        // per-wave h transpose scratch

  const int tid  = threadIdx.x;
  const int lane = tid & 31;
  const int wave = tid >> 5;

  // ---- block-cooperative: pack weight tiles into per-lane bf16 fragments ----
  // Fragment element map (A/B symmetric layout, 16-bit, 16x16x32):
  //   lane l: N = l&15, K-base kb = (l>=16)?8:0
  //   elem j: K = kb + (j<8 ? j : j+8)   (covers K in {kb..kb+7, kb+16..kb+23})
  #pragma unroll
  for (int p = 0; p < 2; ++p) {
    int idx = tid * 2 + p;                    // 0..511 -> (tile, lane)
    int t = idx >> 5, l = idx & 31;
    const float* W = (t < 8) ? Wih : Whh;     // [128,32] row-major
    int row = ((t & 7) << 4) + (l & 15);      // gate index (N dimension)
    int kbw = (l >> 4) ? 8 : 0;
    bf16_t* dst = (bf16_t*)&wfrag[t][l];
    #pragma unroll
    for (int j = 0; j < 16; ++j) {
      int K = kbw + (j < 8 ? j : j + 8);
      dst[j] = (bf16_t)W[row * 32 + K];       // W^T[K][row]
    }
  }
  if (tid < 128) bias[tid] = bih[tid] + bhh[tid];
  __syncthreads();

  const int  hi   = lane >> 4;                // 0 / 1
  const int  ln   = lane & 15;
  const int  kb   = hi ? 8 : 0;
  const long row0 = ((long)blockIdx.x * 8 + wave) * 16;
  if (row0 + 16 > B) return;                  // whole-wave guard

  // ---- A-fragment loader: 16 rows x 32 f32 -> bf16, per-lane K = kb..kb+7, kb+16..kb+23 ----
  auto loadA = [&](const float* __restrict__ base) -> v16bf {
    const float* pr = base + (row0 + ln) * 32;
    v4f a0 = __builtin_nontemporal_load((const v4f*)(pr + kb));
    v4f a1 = __builtin_nontemporal_load((const v4f*)(pr + kb + 4));
    v4f a2 = __builtin_nontemporal_load((const v4f*)(pr + kb + 16));
    v4f a3 = __builtin_nontemporal_load((const v4f*)(pr + kb + 20));
    v16bf a;
    #pragma unroll
    for (int j = 0; j < 4; ++j) {
      a[j]      = (bf16_t)a0[j];
      a[4 + j]  = (bf16_t)a1[j];
      a[8 + j]  = (bf16_t)a2[j];
      a[12 + j] = (bf16_t)a3[j];
    }
    return a;
  };

  // ---- c state in C/D fragment layout: lane holds (M = v+8*hi, N = 16n+ln) ----
  v8f cacc[2];
  #pragma unroll
  for (int n = 0; n < 2; ++n)
    #pragma unroll
    for (int v = 0; v < 8; ++v)
      cacc[n][v] =
          __builtin_nontemporal_load(C0 + (row0 + v + 8 * hi) * 32 + 16 * n + ln);

  // ---- x_gates = X @ W_ih^T + (b_ih + b_hh), kept resident across steps ----
  v16bf Ax = loadA(X);
  v8f xg[8];
  #pragma unroll
  for (int t = 0; t < 8; ++t) {
    float bv = bias[16 * t + ln];             // bias depends only on N
    v8f c;
    #pragma unroll
    for (int v = 0; v < 8; ++v) c[v] = bv;
    xg[t] = wmma_bf16(Ax, wfrag[t][lane], c);
  }

  // ---- 4 recurrent steps ----
  v16bf Ah = loadA(H0);
  v8f hacc[2];
  #pragma unroll
  for (int s = 0; s < 4; ++s) {
    if (s > 0) {
      __syncthreads();                        // scratch stores visible
      const bf16_t* hp = &hscr[wave][ln][0];  // row ln, 80B stride (16B aligned)
      v8bf lo = *(const v8bf*)(hp + kb);      // K = kb .. kb+7
      v8bf hh = *(const v8bf*)(hp + kb + 16); // K = kb+16 .. kb+23
      Ah = __builtin_shufflevector(lo, hh, 0, 1, 2, 3, 4, 5, 6, 7,
                                           8, 9, 10, 11, 12, 13, 14, 15);
    }
    #pragma unroll
    for (int n = 0; n < 2; ++n) {             // torch gate order: i,f,g,o
      v8f gi = wmma_bf16(Ah, wfrag[8 + n][lane],     xg[n]);
      v8f gf = wmma_bf16(Ah, wfrag[8 + n + 2][lane], xg[n + 2]);
      v8f gg = wmma_bf16(Ah, wfrag[8 + n + 4][lane], xg[n + 4]);
      v8f go = wmma_bf16(Ah, wfrag[8 + n + 6][lane], xg[n + 6]);
      #pragma unroll
      for (int v = 0; v < 8; ++v) {
        float c = sigmoidf_fast(gf[v]) * cacc[n][v] +
                  sigmoidf_fast(gi[v]) * fast_tanh(gg[v]);
        cacc[n][v] = c;
        float h = sigmoidf_fast(go[v]) * fast_tanh(c);
        hacc[n][v] = (h >= 0.0f) ? h : NEG_SLOPE * h;   // LeakyReLU
      }
    }
    if (s < 3) {
      // scatter h (D layout) -> row-major bf16 tile for next step's A gather
      #pragma unroll
      for (int n = 0; n < 2; ++n)
        #pragma unroll
        for (int v = 0; v < 8; ++v)
          hscr[wave][v + 8 * hi][16 * n + ln] = (bf16_t)hacc[n][v];
    }
  }

  // ---- store final h, [B,32] f32, coalesced non-temporal ----
  #pragma unroll
  for (int n = 0; n < 2; ++n)
    #pragma unroll
    for (int v = 0; v < 8; ++v)
      __builtin_nontemporal_store(hacc[n][v],
          out + (row0 + v + 8 * hi) * 32 + 16 * n + ln);
}

extern "C" void kernel_launch(void* const* d_in, const int* in_sizes, int n_in,
                              void* d_out, int out_size, void* d_ws, size_t ws_size,
                              hipStream_t stream) {
  const float* X   = (const float*)d_in[0];  // token_embedding [B,32]
  const float* H0  = (const float*)d_in[1];  // hx0 [B,32]
  const float* C0  = (const float*)d_in[2];  // cx0 [B,32]
  const float* Wih = (const float*)d_in[3];  // [128,32]
  const float* Whh = (const float*)d_in[4];  // [128,32]
  const float* bih = (const float*)d_in[5];  // [128]
  const float* bhh = (const float*)d_in[6];  // [128]
  float* out = (float*)d_out;                // h [B,32]

  long B = (long)in_sizes[1] / 32;           // hx0 element count / H
  int  grid = (int)((B + 127) / 128);        // 128 rows per 256-thread block
  lstm4_wmma_kernel<<<grid, 256, 0, stream>>>(X, H0, C0, Wih, Whh, bih, bhh,
                                              out, B);
}
